// MoEEncoderDecoderGPT_18657337934013
// MI455X (gfx1250) — matrix operand order repaired and checked by reading, not compiled, not run on hardware
//
#include <hip/hip_runtime.h>

// ---------------------------------------------------------------------------
// MoE encoder-decoder forward for gfx1250 (MI455X).
// All GEMMs run through v_wmma_f32_16x16x32_f16 (f16 in, f32 accumulate),
// fed from LDS tiles. Epilogues fuse scale / residual / per-row dispatch.
// ---------------------------------------------------------------------------

typedef __attribute__((ext_vector_type(16))) _Float16 v16h;
typedef __attribute__((ext_vector_type(8)))  float    v8f;

// ---------------------------------------------------------------------------
// WMMA GEMM:  C[m][n] = alpha * sum_k A[m][k] * B(k,n)  (+RowScale, +Res)
//   BT=true : Bm is [N][K] row-major (i.e. x @ W.T with W=[N,K])
//   BT=false: Bm is [K][N] row-major (plain A @ B)
// Block: 256 threads = 8 waves; tile 64(M) x 64(N), BK=64.
// Wave w owns a 16x32 strip: M subtile wr=w>>1, N subtiles wc, wc+1.
// A fragment is reused across both WMMAs -> 4 wmma per wave per K-tile.
// Requires M%64==0, N%64==0, K%64==0 (true for every GEMM in this model).
// ---------------------------------------------------------------------------
template<bool BT>
__global__ __launch_bounds__(256) void k_gemm(
    const float* __restrict__ A, int lda,
    const float* __restrict__ Bm, int ldb,
    float* __restrict__ C, int ldc,
    const float* __restrict__ Res, int ldr,
    const float* __restrict__ RowScale, int rss,
    int Kd, float alpha)
{
  constexpr int BK = 64;
  __shared__ _Float16 As[64][BK + 8];
  __shared__ _Float16 Bs[64][BK + 8];

  const int tid  = threadIdx.x;
  const int lane = tid & 31;
  const int w    = tid >> 5;        // wave 0..7
  const int wr   = w >> 1;          // 0..3  (M sub-tile, 16 rows)
  const int wc   = (w & 1) << 1;    // 0 or 2 (first of two N sub-tiles)
  const int half = lane >> 4;       // lane half per ISA fragment layout
  const int lr   = lane & 15;
  const long m0 = (long)blockIdx.y * 64;
  const long n0 = (long)blockIdx.x * 64;

  v8f acc0 = {};
  v8f acc1 = {};

  for (int k0 = 0; k0 < Kd; k0 += BK) {
    // ---- load A tile 64x64 (4096 floats, 4 float4 per thread), f32->f16
    #pragma unroll
    for (int i = 0; i < 4; ++i) {
      int s = tid * 4 + i;
      int r = s >> 4;
      int c = (s & 15) << 2;
      const float4 v = *(const float4*)(A + (m0 + r) * (long)lda + k0 + c);
      As[r][c + 0] = (_Float16)v.x; As[r][c + 1] = (_Float16)v.y;
      As[r][c + 2] = (_Float16)v.z; As[r][c + 3] = (_Float16)v.w;
    }
    // ---- load B tile into Bs[n][k] (4096 floats, 4 float4 per thread)
    #pragma unroll
    for (int i = 0; i < 4; ++i) {
      int s = tid * 4 + i;
      if constexpr (BT) {
        int n = s >> 4;
        int c = (s & 15) << 2;
        const float4 v = *(const float4*)(Bm + (n0 + n) * (long)ldb + k0 + c);
        Bs[n][c + 0] = (_Float16)v.x; Bs[n][c + 1] = (_Float16)v.y;
        Bs[n][c + 2] = (_Float16)v.z; Bs[n][c + 3] = (_Float16)v.w;
      } else {
        int kk = s >> 4;
        int c  = (s & 15) << 2;
        const float4 v = *(const float4*)(Bm + (long)(k0 + kk) * ldb + n0 + c);
        Bs[c + 0][kk] = (_Float16)v.x; Bs[c + 1][kk] = (_Float16)v.y;
        Bs[c + 2][kk] = (_Float16)v.z; Bs[c + 3][kk] = (_Float16)v.w;
      }
    }
    // prefetch next K-tile of A (global_prefetch_b8 path)
    if (k0 + BK < Kd)
      __builtin_prefetch(A + (m0 + (tid >> 2)) * (long)lda + k0 + BK, 0, 1);
    __syncthreads();

    // ---- 4 WMMAs per K-tile: A fragment reused for two B fragments
    #pragma unroll
    for (int kk = 0; kk < BK; kk += 32) {
      v16h a, b0, b1;
      // A fragment (ISA 7.12.2, 16-bit A 16x32): lane half selects K bands
      const _Float16* ap = &As[wr * 16 + lr][kk];
      #pragma unroll
      for (int i = 0; i < 16; ++i) {
        a[i] = ap[half * 16 + i];
      }
      // B fragments: lane half selects K 0-15 / 16-31, contiguous in Bs[n][k]
      const _Float16* bp0 = &Bs[wc * 16 + lr][kk + half * 16];
      const _Float16* bp1 = &Bs[(wc + 1) * 16 + lr][kk + half * 16];
      #pragma unroll
      for (int j = 0; j < 16; ++j) { b0[j] = bp0[j]; b1[j] = bp1[j]; }

      acc0 = __builtin_amdgcn_wmma_f32_16x16x32_f16(false, a, false, b0,
                                                    (short)0, acc0, false, false);
      acc1 = __builtin_amdgcn_wmma_f32_16x16x32_f16(false, a, false, b1,
                                                    (short)0, acc1, false, false);
    }
    __syncthreads();
  }

  // ---- epilogue: C/D layout: VGPR r -> row = half*8 + r, col = lane&15
  const long row0 = m0 + wr * 16 + half * 8;
  const long col0 = n0 + wc * 16 + lr;
  const long col1 = col0 + 16;
  #pragma unroll
  for (int r = 0; r < 8; ++r) {
    long rr = row0 + r;
    float rs = RowScale ? RowScale[rr * (long)rss] : 1.0f;
    float v0 = acc0[r] * alpha * rs;
    float v1 = acc1[r] * alpha * rs;
    if (Res) {
      v0 += Res[rr * (long)ldr + col0];
      v1 += Res[rr * (long)ldr + col1];
    }
    C[rr * (long)ldc + col0] = v0;
    C[rr * (long)ldc + col1] = v1;
  }
}

// ---------------------------------------------------------------------------
// Embedding gather: x[t] = emb[idx[t]] + pos[t % S]
// ---------------------------------------------------------------------------
__global__ __launch_bounds__(256) void k_embed(
    const int* __restrict__ idx, const float* __restrict__ emb,
    const float* __restrict__ pos, float* __restrict__ x)
{
  int t = blockIdx.x;
  int tok = idx[t];
  int s = t & 511;
  for (int d = threadIdx.x; d < 512; d += 256)
    x[(long)t * 512 + d] = emb[(long)tok * 512 + d] + pos[(long)s * 512 + d];
}

// ---------------------------------------------------------------------------
// LayerNorm over D=512, one block per row
// ---------------------------------------------------------------------------
__global__ __launch_bounds__(256) void k_ln(
    const float* __restrict__ in, const float* __restrict__ g,
    const float* __restrict__ b, float* __restrict__ out)
{
  __shared__ float sh[256];
  int row = blockIdx.x, tid = threadIdx.x;
  const float* xp = in + (long)row * 512;
  float s = 0.f;
  for (int d = tid; d < 512; d += 256) s += xp[d];
  sh[tid] = s; __syncthreads();
  for (int off = 128; off > 0; off >>= 1) {
    if (tid < off) sh[tid] += sh[tid + off];
    __syncthreads();
  }
  float mean = sh[0] * (1.f / 512.f);
  __syncthreads();
  s = 0.f;
  for (int d = tid; d < 512; d += 256) { float df = xp[d] - mean; s += df * df; }
  sh[tid] = s; __syncthreads();
  for (int off = 128; off > 0; off >>= 1) {
    if (tid < off) sh[tid] += sh[tid + off];
    __syncthreads();
  }
  float inv = rsqrtf(sh[0] * (1.f / 512.f) + 1e-5f);
  float* op = out + (long)row * 512;
  for (int d = tid; d < 512; d += 256)
    op[d] = (xp[d] - mean) * inv * g[d] + b[d];
}

// ---------------------------------------------------------------------------
// Row softmax over 512 cols with optional causal mask (scale was in GEMM)
// ---------------------------------------------------------------------------
__global__ __launch_bounds__(256) void k_softmax(float* __restrict__ s, int causal)
{
  __shared__ float sh[256];
  int row = blockIdx.x, tid = threadIdx.x;
  int q = row & 511;
  float* p = s + (long)row * 512;
  float v[2];
  float mx = -3.4e38f;
  #pragma unroll
  for (int i = 0; i < 2; ++i) {
    int c = tid + i * 256;
    float t = p[c];
    if (causal && c > q) t = -1e9f;
    v[i] = t;
    mx = fmaxf(mx, t);
  }
  sh[tid] = mx; __syncthreads();
  for (int off = 128; off > 0; off >>= 1) {
    if (tid < off) sh[tid] = fmaxf(sh[tid], sh[tid + off]);
    __syncthreads();
  }
  mx = sh[0]; __syncthreads();
  float sum = 0.f;
  #pragma unroll
  for (int i = 0; i < 2; ++i) { v[i] = __expf(v[i] - mx); sum += v[i]; }
  sh[tid] = sum; __syncthreads();
  for (int off = 128; off > 0; off >>= 1) {
    if (tid < off) sh[tid] += sh[tid + off];
    __syncthreads();
  }
  float inv = 1.f / sh[0];
  #pragma unroll
  for (int i = 0; i < 2; ++i) p[tid + i * 256] = v[i] * inv;
}

// ---------------------------------------------------------------------------
// SiLU gate: hid[t][j] = silu(h[t][2048+j]) * h[t][j]
// ---------------------------------------------------------------------------
__global__ __launch_bounds__(256) void k_silu_gate(
    const float* __restrict__ h, float* __restrict__ hid)
{
  int t = blockIdx.x;
  for (int j = threadIdx.x; j < 2048; j += 256) {
    float x1 = h[(long)t * 4096 + j];
    float x2 = h[(long)t * 4096 + 2048 + j];
    hid[(long)t * 2048 + j] = (x2 / (1.f + __expf(-x2))) * x1;
  }
}

// ---------------------------------------------------------------------------
// Router: logits, softmax, top-2 w/ capacity (per-k snapshot semantics),
// unrouted fallback, normalization, z-loss + load-balance loss into rtot.
// Single block of 1024 threads (one per token).
// ---------------------------------------------------------------------------
__global__ __launch_bounds__(1024) void k_router(
    const float* __restrict__ xn, const float* __restrict__ rw,
    float* __restrict__ dispatch, float* __restrict__ rtot)
{
  const int E = 8;
  const float CAPf = 320.0f;   // int(1.25 * B*S*K/E)
  int t = threadIdx.x;
  __shared__ float counts[E], addc[E], ec[E], zsum;
  __shared__ int least;

  float logits[E];
  const float* xr = xn + (long)t * 512;
  #pragma unroll
  for (int e = 0; e < E; ++e) {
    const float* wr = rw + (long)e * 512;
    float s = 0.f;
    for (int d = 0; d < 512; ++d) s += xr[d] * wr[d];
    logits[e] = s;
  }
  float mx = logits[0];
  #pragma unroll
  for (int e = 1; e < E; ++e) mx = fmaxf(mx, logits[e]);
  float probs[E]; float se = 0.f;
  #pragma unroll
  for (int e = 0; e < E; ++e) { probs[e] = __expf(logits[e] - mx); se += probs[e]; }
  float sinv = 1.f / se;
  #pragma unroll
  for (int e = 0; e < E; ++e) probs[e] *= sinv;

  int i1 = 0; float w1 = probs[0];
  #pragma unroll
  for (int e = 1; e < E; ++e) if (probs[e] > w1) { w1 = probs[e]; i1 = e; }
  int i2 = (i1 == 0) ? 1 : 0; float w2 = probs[i2];
  #pragma unroll
  for (int e = 0; e < E; ++e)
    if (e != i1 && probs[e] > w2) { w2 = probs[e]; i2 = e; }

  float wsum = w1 + w2 + 1e-8f;
  w1 /= wsum; w2 /= wsum;

  float disp[E];
  #pragma unroll
  for (int e = 0; e < E; ++e) disp[e] = 0.f;

  if (t < E) { counts[t] = 0.f; ec[t] = 0.f; }
  if (t == 0) zsum = 0.f;
  __syncthreads();

  int   ei[2] = { i1, i2 };
  float tw[2] = { w1, w2 };
  for (int kk = 0; kk < 2; ++kk) {
    if (t < E) addc[t] = 0.f;
    __syncthreads();
    bool m = counts[ei[kk]] < CAPf;      // snapshot, like reference
    if (m) {
      atomicAdd(&addc[ei[kk]], 1.0f);
      disp[ei[kk]] = tw[kk];
    }
    __syncthreads();
    if (t < E) counts[t] += addc[t];
    __syncthreads();
  }

  float dsum = 0.f;
  #pragma unroll
  for (int e = 0; e < E; ++e) dsum += disp[e];
  if (t == 0) {
    int le = 0; float best = CAPf - counts[0];
    for (int e = 1; e < E; ++e) {
      float r = CAPf - counts[e];
      if (r > best) { best = r; le = e; }
    }
    least = le;
  }
  __syncthreads();
  if (dsum == 0.f) disp[least] = 1.0f;
  dsum = 0.f;
  #pragma unroll
  for (int e = 0; e < E; ++e) dsum += disp[e];
  float inv = 1.f / (dsum + 1e-8f);
  float z = 0.f;
  #pragma unroll
  for (int e = 0; e < E; ++e) {
    disp[e] *= inv;
    dispatch[(long)t * E + e] = disp[e];
    atomicAdd(&ec[e], disp[e]);
    z += logits[e] * logits[e];
  }
  atomicAdd(&zsum, z);
  __syncthreads();
  if (t == 0) {
    float zl = zsum * (1.f / (1024.f * 8.f));
    float lb = 0.f;
    for (int e = 0; e < E; ++e) {
      float d = ec[e] * (1.f / 1024.f) - 256.f / 1024.f;
      lb += d * d;
    }
    lb *= (1.f / 8.f);
    atomicAdd(rtot, 0.001f * zl + 0.001f * lb);
  }
}

// ---------------------------------------------------------------------------
// Host orchestration
// ---------------------------------------------------------------------------
extern "C" void kernel_launch(void* const* d_in, const int* in_sizes, int n_in,
                              void* d_out, int out_size, void* d_ws, size_t ws_size,
                              hipStream_t stream) {
  (void)in_sizes; (void)n_in; (void)out_size; (void)ws_size;

  const int T = 1024;            // B*S
  const long TD = (long)T * 512;

  auto F = [&](int i) { return (const float*)d_in[i]; };
  const int* eidx = (const int*)d_in[0];
  const int* didx = (const int*)d_in[1];
  const float* emb = F(2);
  const float* pos = F(3);

  // ---- workspace carve (floats) ----
  float* ws = (float*)d_ws;
  float* x0      = ws;               ws += TD;
  float* x1      = ws;               ws += TD;
  float* xn      = ws;               ws += TD;
  float* xn2     = ws;               ws += TD;
  float* attno   = ws;               ws += TD;
  float* enc_out = ws;               ws += TD;
  float* qkv     = ws;               ws += (long)T * 1536;
  float* scores  = ws;               ws += (long)2 * 8 * 512 * 512;  // aliased w/ hbuf
  float* hbuf    = scores;                                           // 1024*4096
  float* hid     = ws;               ws += (long)T * 2048;
  float* disp    = ws;               ws += (long)T * 8;

  float* logits_out = (float*)d_out;
  float* rtot = logits_out + (long)T * 32000;
  hipMemsetAsync(rtot, 0, sizeof(float), stream);

  auto gemmT = [&](const float* A, int lda, const float* Bm, int ldb,
                   float* C, int ldc, const float* Res, int ldr,
                   const float* Rs, int rss, int M, int N, int K, float alpha) {
    dim3 g(N / 64, M / 64);
    k_gemm<true><<<g, 256, 0, stream>>>(A, lda, Bm, ldb, C, ldc, Res, ldr, Rs, rss, K, alpha);
  };
  auto gemmN = [&](const float* A, int lda, const float* Bm, int ldb,
                   float* C, int ldc, int M, int N, int K, float alpha) {
    dim3 g(N / 64, M / 64);
    k_gemm<false><<<g, 256, 0, stream>>>(A, lda, Bm, ldb, C, ldc,
                                         nullptr, 0, nullptr, 0, K, alpha);
  };

  auto attn_core = [&](const float* qkvBuf, int causal) {
    for (int b = 0; b < 2; ++b)
      for (int h = 0; h < 8; ++h) {
        const float* Q  = qkvBuf + (long)b * 512 * 1536 + h * 64;
        const float* Kp = qkvBuf + (long)b * 512 * 1536 + 512 + h * 64;
        float* sc = scores + ((long)(b * 8 + h)) * 512 * 512;
        gemmT(Q, 1536, Kp, 1536, sc, 512, nullptr, 0, nullptr, 0,
              512, 512, 64, 0.125f);  // 1/sqrt(64)
      }
    k_softmax<<<8192, 256, 0, stream>>>(scores, causal);
    for (int b = 0; b < 2; ++b)
      for (int h = 0; h < 8; ++h) {
        const float* Vp = qkvBuf + (long)b * 512 * 1536 + 1024 + h * 64;
        const float* sc = scores + ((long)(b * 8 + h)) * 512 * 512;
        gemmN(sc, 512, Vp, 1536, attno + (long)b * 512 * 512 + h * 64, 512,
              512, 64, 512, 1.0f);
      }
  };

  auto self_attn = [&](float* x, const float* g, const float* b,
                       const float* wqkv, const float* wo, int causal) {
    k_ln<<<T, 256, 0, stream>>>(x, g, b, xn);
    gemmT(xn, 512, wqkv, 512, qkv, 1536, nullptr, 0, nullptr, 0,
          T, 1536, 512, 1.0f);
    attn_core(qkv, causal);
    gemmT(attno, 512, wo, 512, x, 512, /*Res=*/x, 512, nullptr, 0,
          T, 512, 512, 1.0f);
  };

  auto moe = [&](float*& x, float*& xalt,
                 const float* ln2g, const float* ln2b,
                 const float* mng, const float* mnb,
                 const float* rw, const float* gu, const float* dn) {
    k_ln<<<T, 256, 0, stream>>>(x, ln2g, ln2b, xn);
    k_ln<<<T, 256, 0, stream>>>(xn, mng, mnb, xn2);   // double-LN per reference
    k_router<<<1, 1024, 0, stream>>>(xn2, rw, disp, rtot);
    hipMemcpyAsync(xalt, x, TD * sizeof(float), hipMemcpyDeviceToDevice, stream);
    for (int e = 0; e < 8; ++e) {
      gemmT(xn2, 512, gu + (long)e * 4096 * 512, 512, hbuf, 4096,
            nullptr, 0, nullptr, 0, T, 4096, 512, 1.0f);
      k_silu_gate<<<T, 256, 0, stream>>>(hbuf, hid);
      // xalt += dispatch[:,e] * (hid @ dn[e].T)
      gemmT(hid, 2048, dn + (long)e * 512 * 2048, 2048, xalt, 512,
            /*Res=*/xalt, 512, /*RowScale=*/disp + e, 8, T, 512, 2048, 1.0f);
    }
    float* tmp = x; x = xalt; xalt = tmp;
  };

  float* x = x0; float* xalt = x1;

  // ---- encoder ----
  k_embed<<<T, 256, 0, stream>>>(eidx, emb, pos, x);
  for (int l = 0; l < 2; ++l) {
    int b0 = 4 + l * 11;
    self_attn(x, F(b0 + 0), F(b0 + 1), F(b0 + 2), F(b0 + 3), 1);
    moe(x, xalt, F(b0 + 4), F(b0 + 5), F(b0 + 6), F(b0 + 7),
        F(b0 + 8), F(b0 + 9), F(b0 + 10));
  }
  k_ln<<<T, 256, 0, stream>>>(x, F(56), F(57), enc_out);

  // ---- decoder ----
  k_embed<<<T, 256, 0, stream>>>(didx, emb, pos, x);
  for (int l = 0; l < 2; ++l) {
    int b0 = 26 + l * 11;
    int c0 = 48 + l * 4;
    self_attn(x, F(b0 + 0), F(b0 + 1), F(b0 + 2), F(b0 + 3), 1);
    moe(x, xalt, F(b0 + 4), F(b0 + 5), F(b0 + 6), F(b0 + 7),
        F(b0 + 8), F(b0 + 9), F(b0 + 10));
    // cross attention: q from LN(y), k/v from enc_out, non-causal
    k_ln<<<T, 256, 0, stream>>>(x, F(c0 + 2), F(c0 + 3), xn);
    gemmT(xn, 512, F(c0 + 0), 512, qkv, 1536, nullptr, 0, nullptr, 0,
          T, 512, 512, 1.0f);
    gemmT(enc_out, 512, F(c0 + 0) + (long)512 * 512, 512, qkv + 512, 1536,
          nullptr, 0, nullptr, 0, T, 1024, 512, 1.0f);
    attn_core(qkv, 0);
    gemmT(attno, 512, F(c0 + 1), 512, x, 512, /*Res=*/x, 512, nullptr, 0,
          T, 512, 512, 1.0f);
  }

  // ---- final LN + vocab projection ----
  k_ln<<<T, 256, 0, stream>>>(x, F(58), F(59), xn);
  gemmT(xn, 512, emb, 512, logits_out, 32000, nullptr, 0, nullptr, 0,
        T, 32000, 512, 1.0f);
}
